// GroupedQueryAttention_85684597555660
// MI455X (gfx1250) — matrix operand order repaired and checked by reading, not compiled, not run on hardware
//
#include <hip/hip_runtime.h>
#include <hip/hip_bf16.h>

typedef _Float16 half_t;
typedef __attribute__((ext_vector_type(16))) _Float16 v16h;
typedef __attribute__((ext_vector_type(8)))  _Float16 v8h;
typedef __attribute__((ext_vector_type(8)))  float    v8f;

#define WMMA_F32_F16(a, b, c) \
  __builtin_amdgcn_wmma_f32_16x16x32_f16(false, (a), false, (b), (short)0, (c), false, false)

static __device__ __forceinline__ v16h pack16(v8h lo, v8h hi) {
  v16h r;
#pragma unroll
  for (int i = 0; i < 8; ++i) { r[i] = lo[i]; r[i + 8] = hi[i]; }
  return r;
}

static __device__ __forceinline__ v8f zero8() {
  v8f z;
#pragma unroll
  for (int i = 0; i < 8; ++i) z[i] = 0.0f;
  return z;
}

// ---- 16-lane row reductions: DPP row_ror butterfly (VALU pipe, no LDS) ----
#if __has_builtin(__builtin_amdgcn_mov_dpp)
#define ROR_F32(x, n) \
  __int_as_float(__builtin_amdgcn_mov_dpp(__float_as_int(x), 0x120 | (n), 0xF, 0xF, 1))
static __device__ __forceinline__ float rowmax16(float x) {
  x = fmaxf(x, ROR_F32(x, 8));
  x = fmaxf(x, ROR_F32(x, 4));
  x = fmaxf(x, ROR_F32(x, 2));
  x = fmaxf(x, ROR_F32(x, 1));
  return x;
}
static __device__ __forceinline__ float rowsum16(float x) {
  x += ROR_F32(x, 8);
  x += ROR_F32(x, 4);
  x += ROR_F32(x, 2);
  x += ROR_F32(x, 1);
  return x;
}
#else
static __device__ __forceinline__ float rowmax16(float x) {
  for (int off = 8; off; off >>= 1) x = fmaxf(x, __shfl_xor(x, off, 32));
  return x;
}
static __device__ __forceinline__ float rowsum16(float x) {
  for (int off = 8; off; off >>= 1) x += __shfl_xor(x, off, 32);
  return x;
}
#endif

// ---------------------------------------------------------------------------
// Tiled WMMA GEMM: Out[M,N] = alpha * A[M,K] * B[N,K]^T + bias[N]
// Double-buffered LDS (ping-pong, one barrier per K-step), f32->f16 convert
// in the tile fill, global_prefetch two tiles ahead.
// Block: 256 threads = 8 waves; tile 128x128; K-step 32.
// ---------------------------------------------------------------------------
template <bool A_HALF, bool OUT_FLOAT>
__global__ __launch_bounds__(256) void gemm_wmma_kernel(
    const void* __restrict__ Ap, const float* __restrict__ B,
    const float* __restrict__ bias, void* __restrict__ Outp,
    int M, int N, int K, float alpha) {
  __shared__ __attribute__((aligned(16))) half_t As[2][128 * 40];  // stride 40 halves = 80B
  __shared__ __attribute__((aligned(16))) half_t Bs[2][128 * 40];

  const int tid   = threadIdx.x;
  const int lane  = tid & 31;
  const int wave  = tid >> 5;
  const int waveM = wave & 3;
  const int waveN = wave >> 2;
  const int lrow  = lane & 15;
  const int kh    = lane >> 4;
  const int bM    = blockIdx.y * 128;
  const int bN    = blockIdx.x * 128;

  v8f acc[2][4];
#pragma unroll
  for (int mi = 0; mi < 2; ++mi)
#pragma unroll
    for (int ni = 0; ni < 4; ++ni) acc[mi][ni] = zero8();

  const int ldRow = tid >> 1;        // 0..127
  const int ldK   = (tid & 1) * 16;  // 0 or 16

  auto loadA = [&](int k0, v8h& h0, v8h& h1) {
    if (A_HALF) {
      const half_t* p = (const half_t*)Ap + (size_t)(bM + ldRow) * K + k0 + ldK;
      h0 = *(const v8h*)p;
      h1 = *(const v8h*)(p + 8);
    } else {
      const float* p = (const float*)Ap + (size_t)(bM + ldRow) * K + k0 + ldK;
#pragma unroll
      for (int i = 0; i < 8; ++i) { h0[i] = (half_t)p[i]; h1[i] = (half_t)p[i + 8]; }
    }
  };
  auto loadB = [&](int k0, v8h& h0, v8h& h1) {
    const float* p = B + (size_t)(bN + ldRow) * K + k0 + ldK;
#pragma unroll
    for (int i = 0; i < 8; ++i) { h0[i] = (half_t)p[i]; h1[i] = (half_t)p[i + 8]; }
  };

  // prologue: tile 0 into buffer 0
  v8h a0, a1, b0, b1;
  loadA(0, a0, a1);
  loadB(0, b0, b1);
  *(v8h*)(As[0] + ldRow * 40 + ldK)     = a0;
  *(v8h*)(As[0] + ldRow * 40 + ldK + 8) = a1;
  *(v8h*)(Bs[0] + ldRow * 40 + ldK)     = b0;
  *(v8h*)(Bs[0] + ldRow * 40 + ldK + 8) = b1;
  __syncthreads();

  for (int k0 = 0; k0 < K; k0 += 32) {
    const int cur = (k0 >> 5) & 1;
    const bool more = (k0 + 32) < K;

    // issue next tile's global loads before compute (latency overlap)
    if (more) {
      loadA(k0 + 32, a0, a1);
      loadB(k0 + 32, b0, b1);
      if (k0 + 64 < K) {  // prefetch two tiles ahead -> global_prefetch_b8
        if (A_HALF)
          __builtin_prefetch((const half_t*)Ap + (size_t)(bM + ldRow) * K + k0 + 64 + ldK, 0, 1);
        else
          __builtin_prefetch((const float*)Ap + (size_t)(bM + ldRow) * K + k0 + 64 + ldK, 0, 1);
        __builtin_prefetch(B + (size_t)(bN + ldRow) * K + k0 + 64 + ldK, 0, 1);
      }
    }

    // ---- fragments from current buffer + 8 WMMAs ----
    v16h af[2];
#pragma unroll
    for (int mi = 0; mi < 2; ++mi) {
      const half_t* pa = As[cur] + (waveM * 32 + mi * 16 + lrow) * 40 + kh * 8;
      af[mi] = pack16(*(const v8h*)pa, *(const v8h*)(pa + 16));
    }
    v16h bf[4];
#pragma unroll
    for (int ni = 0; ni < 4; ++ni) {
      const half_t* pb = Bs[cur] + (waveN * 64 + ni * 16 + lrow) * 40 + kh * 8;
      bf[ni] = pack16(*(const v8h*)pb, *(const v8h*)(pb + 16));
    }
#pragma unroll
    for (int mi = 0; mi < 2; ++mi)
#pragma unroll
      for (int ni = 0; ni < 4; ++ni) acc[mi][ni] = WMMA_F32_F16(af[mi], bf[ni], acc[mi][ni]);

    // store next tile into the other buffer (others still read cur: safe)
    if (more) {
      half_t* as = As[cur ^ 1];
      half_t* bs = Bs[cur ^ 1];
      *(v8h*)(as + ldRow * 40 + ldK)     = a0;
      *(v8h*)(as + ldRow * 40 + ldK + 8) = a1;
      *(v8h*)(bs + ldRow * 40 + ldK)     = b0;
      *(v8h*)(bs + ldRow * 40 + ldK + 8) = b1;
    }
    __syncthreads();
  }

  // ---- epilogue: C layout row = v + kh*8, col = lrow ----
#pragma unroll
  for (int mi = 0; mi < 2; ++mi) {
#pragma unroll
    for (int ni = 0; ni < 4; ++ni) {
      const int col = bN + waveN * 64 + ni * 16 + lrow;
      const float bi = bias ? bias[col] : 0.0f;
#pragma unroll
      for (int v = 0; v < 8; ++v) {
        const int row = bM + waveM * 32 + mi * 16 + kh * 8 + v;
        const float val = acc[mi][ni][v] * alpha + bi;
        if (OUT_FLOAT) ((float*)Outp)[(size_t)row * N + col] = val;
        else           ((half_t*)Outp)[(size_t)row * N + col] = (half_t)val;
      }
    }
  }
}

// ---------------------------------------------------------------------------
// Flash attention: one head per blockIdx.y, 64 query rows per block (4 waves
// x 16 rows), S=2048 keys in chunks of 32, online softmax, WMMA everywhere.
// K chunk copied memory->LDS with GLOBAL_LOAD_ASYNC_TO_LDS_B128 (ASYNCcnt).
// ---------------------------------------------------------------------------
__global__ __launch_bounds__(128) void attn_kernel(
    const half_t* __restrict__ Qh, const half_t* __restrict__ Kh,
    const half_t* __restrict__ Vh, half_t* __restrict__ Ch) {
  __shared__ __attribute__((aligned(16))) half_t Ks[32 * 136];   // [key][d]
  __shared__ __attribute__((aligned(16))) half_t Vt[128 * 40];   // [d][key]
  __shared__ __attribute__((aligned(16))) half_t Ps[4][16 * 40]; // per-wave probs

  const int tid  = threadIdx.x;
  const int lane = tid & 31;
  const int wave = tid >> 5;
  const int lrow = lane & 15;
  const int kh   = lane >> 4;
  const int head = blockIdx.y;
  const int kvh  = head >> 2;  // G = 4
  const int qBase = blockIdx.x * 64 + wave * 16;

  v16h qf[4];
  {
    const half_t* qr = Qh + (size_t)(qBase + lrow) * 4096 + head * 128;
#pragma unroll
    for (int kc = 0; kc < 4; ++kc) {
      v8h lo = *(const v8h*)(qr + kc * 32 + kh * 8);
      v8h hi = *(const v8h*)(qr + kc * 32 + 16 + kh * 8);
      qf[kc] = pack16(lo, hi);
    }
  }

  v8f ctx[8];
#pragma unroll
  for (int nb = 0; nb < 8; ++nb) ctx[nb] = zero8();
  float m_i[8], l_i[8];
#pragma unroll
  for (int v = 0; v < 8; ++v) { m_i[v] = -1e30f; l_i[v] = 0.0f; }

  const int ldr = tid >> 2;        // key row 0..31
  const int ldc = (tid & 3) * 32;  // d offset 0,32,64,96

  for (int kb = 0; kb < 64; ++kb) {
    {
      // K chunk: async DMA memory -> LDS (no VGPR round-trip)
      const half_t* ksrc = Kh + (size_t)(kb * 32 + ldr) * 1024 + kvh * 128 + ldc;
      half_t* kdst = Ks + ldr * 136 + ldc;
#pragma unroll
      for (int j = 0; j < 32; j += 8) {
        const uint32_t laddr = (uint32_t)(uintptr_t)(kdst + j);
        const uint64_t gaddr = (uint64_t)(uintptr_t)(ksrc + j);
        asm volatile("global_load_async_to_lds_b128 %0, %1, off"
                     :: "v"(laddr), "v"(gaddr) : "memory");
      }
      // V chunk transposed into Vt (needs lane transpose, so via VGPRs)
      const half_t* vsrc = Vh + (size_t)(kb * 32 + ldr) * 1024 + kvh * 128 + ldc;
#pragma unroll
      for (int j0 = 0; j0 < 32; j0 += 8) {
        v8h x = *(const v8h*)(vsrc + j0);
#pragma unroll
        for (int j = 0; j < 8; ++j) Vt[(ldc + j0 + j) * 40 + ldr] = x[j];
      }
      asm volatile("s_wait_asynccnt 0" ::: "memory");
    }
    __syncthreads();

    // scores: two 16x16 key blocks, K-dim = D = 128 (4 x WMMA-K32 each)
    v8f s0 = zero8(), s1 = zero8();
#pragma unroll
    for (int kc = 0; kc < 4; ++kc) {
      const half_t* b0p = Ks + lrow * 136 + kc * 32 + kh * 8;
      const half_t* b1p = Ks + (16 + lrow) * 136 + kc * 32 + kh * 8;
      v16h b0 = pack16(*(const v8h*)b0p, *(const v8h*)(b0p + 16));
      v16h b1 = pack16(*(const v8h*)b1p, *(const v8h*)(b1p + 16));
      s0 = WMMA_F32_F16(qf[kc], b0, s0);
      s1 = WMMA_F32_F16(qf[kc], b1, s1);
    }

    // online softmax (rows live across 16-lane half-waves; DPP reductions)
#pragma unroll
    for (int v = 0; v < 8; ++v) {
      const float mx = rowmax16(fmaxf(s0[v], s1[v]));
      const float mn = fmaxf(m_i[v], mx);
      const float corr = __expf(m_i[v] - mn);
      const float p0 = __expf(s0[v] - mn);
      const float p1 = __expf(s1[v] - mn);
      const float ps = rowsum16(p0 + p1);
      l_i[v] = l_i[v] * corr + ps;
      m_i[v] = mn;
      s0[v] = p0; s1[v] = p1;
#pragma unroll
      for (int nb = 0; nb < 8; ++nb) ctx[nb][v] *= corr;
    }

    // lane-transpose probs via per-wave LDS buffer (C layout -> A layout)
    half_t* P = &Ps[wave][0];
#pragma unroll
    for (int v = 0; v < 8; ++v) {
      const int row = v + kh * 8;
      P[row * 40 + lrow]      = (half_t)s0[v];
      P[row * 40 + 16 + lrow] = (half_t)s1[v];
    }
    __asm__ volatile("s_wait_dscnt 0" ::: "memory");  // wave-internal LDS RAW

    v16h pf;
    {
      const half_t* pp = P + lrow * 40 + kh * 8;
      pf = pack16(*(const v8h*)pp, *(const v8h*)(pp + 16));
    }

    // ctx += probs(16x32) @ V(32x128): 8 N-blocks of 16
#pragma unroll
    for (int nb = 0; nb < 8; ++nb) {
      const half_t* vp = Vt + (nb * 16 + lrow) * 40 + kh * 8;
      v16h bf = pack16(*(const v8h*)vp, *(const v8h*)(vp + 16));
      ctx[nb] = WMMA_F32_F16(pf, bf, ctx[nb]);
    }
    __syncthreads();  // protect Ks/Vt before next chunk overwrites
  }

  // normalize and store ctx (f16) in [s, head*128 + d] layout
#pragma unroll
  for (int v = 0; v < 8; ++v) {
    const float inv = 1.0f / l_i[v];
    const int row = qBase + v + kh * 8;
#pragma unroll
    for (int nb = 0; nb < 8; ++nb) {
      Ch[(size_t)row * 4096 + head * 128 + nb * 16 + lrow] = (half_t)(ctx[nb][v] * inv);
    }
  }
}

// ---------------------------------------------------------------------------
extern "C" void kernel_launch(void* const* d_in, const int* in_sizes, int n_in,
                              void* d_out, int out_size, void* d_ws, size_t ws_size,
                              hipStream_t stream) {
  const float* hidden = (const float*)d_in[0];
  const float* Wq     = (const float*)d_in[1];
  const float* Wk     = (const float*)d_in[2];
  const float* Wv     = (const float*)d_in[3];
  const float* bv     = (const float*)d_in[4];
  const float* Wo     = (const float*)d_in[5];
  float* out = (float*)d_out;

  const int S = 2048, H = 4096, KV = 1024;

  half_t* Qh = (half_t*)d_ws;        // [S][H]  f16, pre-scaled by 1/sqrt(128)
  half_t* Kh = Qh + (size_t)S * H;   // [S][KV] f16
  half_t* Vh = Kh + (size_t)S * KV;  // [S][KV] f16 (bias applied)
  half_t* Ch = Vh + (size_t)S * KV;  // [S][H]  f16 attention context

  dim3 blk(256);
  const float qscale = 0.08838834764831845f;  // 1/sqrt(128)

  gemm_wmma_kernel<false, false><<<dim3(H / 128, S / 128), blk, 0, stream>>>(
      hidden, Wq, nullptr, Qh, S, H, H, qscale);
  gemm_wmma_kernel<false, false><<<dim3(KV / 128, S / 128), blk, 0, stream>>>(
      hidden, Wk, nullptr, Kh, S, KV, H, 1.0f);
  gemm_wmma_kernel<false, false><<<dim3(KV / 128, S / 128), blk, 0, stream>>>(
      hidden, Wv, bv, Vh, S, KV, H, 1.0f);

  attn_kernel<<<dim3(S / 64, 32), dim3(128), 0, stream>>>(Qh, Kh, Vh, Ch);

  gemm_wmma_kernel<true, true><<<dim3(H / 128, S / 128), blk, 0, stream>>>(
      Ch, Wo, nullptr, out, S, H, H, 1.0f);
}
